// SwiGLUMoE_68418829025888
// MI455X (gfx1250) — matrix operand order repaired and checked by reading, not compiled, not run on hardware
//
#include <hip/hip_runtime.h>
#include <hip/hip_bf16.h>
#include <cstdint>
#include <cstddef>

// ---- problem constants (match reference) ----
#define HDIM 2048
#define IDIM 1408
#define NEXP 8
#define NTOK 4096   // B*S = 2*2048

// ---- GEMM tiling ----
#define TM 64       // block M (tokens)
#define TN 64       // stage A block N
#define TNB 128     // stage B block N
#define TKK 32      // block K (bf16 wmma K=32)

// CDNA5 async global->LDS copy path (guarded; falls back to reg pipeline)
#if defined(__AMDGCN__) && __has_builtin(__builtin_amdgcn_global_load_async_to_lds_b128) && __has_builtin(__builtin_amdgcn_s_wait_asynccnt)
#define HAVE_ASYNC_LDS 1
#else
#define HAVE_ASYNC_LDS 0
#endif

typedef __attribute__((ext_vector_type(16))) __bf16 v16bf;
typedef __attribute__((ext_vector_type(8)))  float  v8f;
typedef __attribute__((ext_vector_type(2)))  __bf16 v2bf;
typedef __attribute__((ext_vector_type(2)))  float  v2f;

#if HAVE_ASYNC_LDS
typedef __attribute__((ext_vector_type(4))) int v4i_;
// async copy operands: global side = AS(1) v4i*, LDS side = AS(3) v4i*
#define ASYNC_GPTR(p) ((__attribute__((address_space(1))) v4i_*)(p))
#define ASYNC_LPTR(p) ((__attribute__((address_space(3))) v4i_*)(p))
#endif

union Frag { v16bf v; uint4 q[2]; };

__device__ __forceinline__ v8f wmma_bf16(const Frag& a, const Frag& b, v8f c) {
  // D = A(16x32) * B(32x16) + C(16x16 f32)
  return __builtin_amdgcn_wmma_f32_16x16x32_bf16(false, a.v, false, b.v,
                                                 (short)0, c, false, false);
}

// pack two f32 -> packed bf16 pair; vector conversion lowers to a single
// v_cvt_pk_bf16_f32 (the scalar union path costs ~4 VALU per pair)
__device__ __forceinline__ unsigned pkbf(float a, float b) {
  v2f f; f.x = a; f.y = b;
  v2bf r = __builtin_convertvector(f, v2bf);
  return __builtin_bit_cast(unsigned, r);
}

// Split workgroup barrier that only drains LDS traffic (DScnt), NOT the
// wave-private global register-prefetch loads (__syncthreads forces
// s_wait_loadcnt 0 too, which exposes the prefetch latency at the barrier).
__device__ __forceinline__ void block_sync_lds() {
#if defined(__AMDGCN__)
  asm volatile("s_wait_dscnt 0x0\n\t"
               "s_barrier_signal -1\n\t"
               "s_barrier_wait -1" ::: "memory");
#else
  __syncthreads();
#endif
}

// ---------------------------------------------------------------------------
// 1) zero output accumulator + per-expert counters (ws is NOT re-poisoned)
// ---------------------------------------------------------------------------
__global__ __launch_bounds__(256) void zero_init_kernel(float* __restrict__ out,
                                                        int n, int* __restrict__ counts) {
  int i = blockIdx.x * blockDim.x + threadIdx.x;
#pragma unroll
  for (int j = 0; j < 4; ++j) {
    int k = i * 4 + j;
    if (k < n) out[k] = 0.0f;
  }
  if (i < NEXP) counts[i] = 0;
}

// ---------------------------------------------------------------------------
// 2) x (fp32) -> bf16 activations, once
// ---------------------------------------------------------------------------
__global__ __launch_bounds__(256) void cvt_x_kernel(const float* __restrict__ x,
                                                    __bf16* __restrict__ xb) {
  int i = (blockIdx.x * blockDim.x + threadIdx.x) * 4;
  float4 v = *(const float4*)(x + i);
  unsigned lo = pkbf(v.x, v.y);
  unsigned hi = pkbf(v.z, v.w);
  *(uint2*)(xb + i) = make_uint2(lo, hi);
}

// ---------------------------------------------------------------------------
// 3) router: logits -> softmax -> top2 -> renorm -> compact per-expert lists
//    one wave (32 lanes) per token
// ---------------------------------------------------------------------------
__global__ __launch_bounds__(256) void router_kernel(const float* __restrict__ x,
                                                     const float* __restrict__ Wr,
                                                     int* __restrict__ counts,
                                                     int* __restrict__ tok_idx,
                                                     float* __restrict__ tok_w) {
  const int wave = threadIdx.x >> 5;
  const int lane = threadIdx.x & 31;
  const int t = blockIdx.x * 8 + wave;

  float acc[NEXP];
#pragma unroll
  for (int e = 0; e < NEXP; ++e) acc[e] = 0.0f;

  const float* xr = x + (size_t)t * HDIM;
  for (int h = lane; h < HDIM; h += 32) {
    float xv = xr[h];
    const float4* w4 = (const float4*)(Wr + (size_t)h * NEXP);
    float4 a = w4[0], b = w4[1];
    acc[0] += xv * a.x; acc[1] += xv * a.y; acc[2] += xv * a.z; acc[3] += xv * a.w;
    acc[4] += xv * b.x; acc[5] += xv * b.y; acc[6] += xv * b.z; acc[7] += xv * b.w;
  }
#pragma unroll
  for (int off = 16; off >= 1; off >>= 1) {
#pragma unroll
    for (int e = 0; e < NEXP; ++e) acc[e] += __shfl_xor(acc[e], off, 32);
  }

  if (lane == 0) {
    float m = acc[0];
#pragma unroll
    for (int e = 1; e < NEXP; ++e) m = fmaxf(m, acc[e]);
    float p[NEXP];
#pragma unroll
    for (int e = 0; e < NEXP; ++e) p[e] = __expf(acc[e] - m);
    int i0 = 0;
#pragma unroll
    for (int e = 1; e < NEXP; ++e) if (p[e] > p[i0]) i0 = e;
    int i1 = (i0 == 0) ? 1 : 0;
#pragma unroll
    for (int e = 0; e < NEXP; ++e) if (e != i0 && p[e] > p[i1]) i1 = e;
    float inv = 1.0f / (p[i0] + p[i1]);        // softmax denom cancels
    float w0 = p[i0] * inv, w1 = p[i1] * inv;
    int s0 = atomicAdd(&counts[i0], 1);
    tok_idx[i0 * NTOK + s0] = t;  tok_w[i0 * NTOK + s0] = w0;
    int s1 = atomicAdd(&counts[i1], 1);
    tok_idx[i1 * NTOK + s1] = t;  tok_w[i1 * NTOK + s1] = w1;
  }
}

// ---------------------------------------------------------------------------
// 4) gate+up grouped GEMM + fused SiLU*up -> hbuf (bf16), per expert (grid.z)
//    block 64x64, 8 waves = 2(M) x 4(N); wave = 32x16 for BOTH gate and up.
//    A tile: async global->LDS (double-buffered) or reg-pipelined copy.
//    B tiles: reg-prefetched fp32, 4x4 transpose-convert, ds_store_b64.
// ---------------------------------------------------------------------------
__global__ __launch_bounds__(256) void expert_gateup_kernel(
    const __bf16* __restrict__ xb, const float* __restrict__ Wg,
    const float* __restrict__ Wu, const int* __restrict__ counts,
    const int* __restrict__ tok_idx, __bf16* __restrict__ hbuf) {
  const int e   = blockIdx.z;
  const int cnt = counts[e];
  const int m0  = blockIdx.x * TM;
  if (m0 >= cnt) return;
  const int n0  = blockIdx.y * TN;

  __shared__ __align__(16) __bf16 As[2][TM * TKK];  // double buffer (async path)
  __shared__ __align__(16) __bf16 Bg[TN * TKK];     // [n][k] transposed
  __shared__ __align__(16) __bf16 Bu[TN * TKK];

  const int tid  = threadIdx.x;
  const int lane = tid & 31;
  const int wave = tid >> 5;
  const int wm   = (wave >> 2) << 5;
  const int wn   = (wave & 3) << 4;
  const bool hi  = lane >= 16;
  const int l16  = lane & 15;

  // A fill: 256 threads x 16B, gathered token rows
  const int arow  = tid >> 2;
  const int acol  = (tid & 3) << 3;
  const int aslot = m0 + arow;
  const int atok  = tok_idx[e * NTOK + (aslot < cnt ? aslot : 0)];
  const __bf16* asrc = xb + (size_t)atok * HDIM + acol;

  const float* wgB = Wg + (size_t)e * HDIM * IDIM + n0;
  const float* wuB = Wu + (size_t)e * HDIM * IDIM + n0;

  // B fill: threads 0-127 -> gate, 128-255 -> up; 4(k)x4(n) sub-block each
  const float* wB = (tid & 128) ? wuB : wgB;
  __bf16*      Bt = (tid & 128) ? Bu  : Bg;
  const int sub = tid & 127;
  const int kb  = (sub >> 4) << 2;   // 0..28
  const int nb  = (sub & 15) << 2;   // 0..60

  float4 br0, br1, br2, br3;
#if !HAVE_ASYNC_LDS
  uint4 aReg;
#endif

  v8f cg0 = {}, cg1 = {}, cu0 = {}, cu1 = {};
  const int NIT = HDIM / TKK;

  {  // prologue prefetch (k0 = 0)
    const float* p = wB + (size_t)kb * IDIM + nb;
    br0 = *(const float4*)p;              br1 = *(const float4*)(p + IDIM);
    br2 = *(const float4*)(p + 2 * IDIM); br3 = *(const float4*)(p + 3 * IDIM);
  }
#if HAVE_ASYNC_LDS
  __builtin_amdgcn_global_load_async_to_lds_b128(
      ASYNC_GPTR(asrc), ASYNC_LPTR(&As[0][arow * TKK + acol]), 0, 0);
#else
  aReg = *(const uint4*)asrc;
#endif

  for (int it = 0; it < NIT; ++it) {
    const int k0 = it * TKK;
#if HAVE_ASYNC_LDS
    if (it + 1 < NIT)
      __builtin_amdgcn_global_load_async_to_lds_b128(
          ASYNC_GPTR(asrc + k0 + TKK),
          ASYNC_LPTR(&As[(it + 1) & 1][arow * TKK + acol]), 0, 0);
#else
    *(uint4*)&As[0][arow * TKK + acol] = aReg;
#endif
    {  // commit B regs -> LDS (transpose + cvt, 8B stores)
      const float* f0 = (const float*)&br0;
      const float* f1 = (const float*)&br1;
      const float* f2 = (const float*)&br2;
      const float* f3 = (const float*)&br3;
#pragma unroll
      for (int j = 0; j < 4; ++j) {
        uint2 v; v.x = pkbf(f0[j], f1[j]); v.y = pkbf(f2[j], f3[j]);
        *(uint2*)&Bt[(nb + j) * TKK + kb] = v;
      }
    }
    if (it + 1 < NIT) {  // prefetch next slab into regs (overlaps with wmma)
      const float* p = wB + (size_t)(k0 + TKK + kb) * IDIM + nb;
      br0 = *(const float4*)p;              br1 = *(const float4*)(p + IDIM);
      br2 = *(const float4*)(p + 2 * IDIM); br3 = *(const float4*)(p + 3 * IDIM);
#if !HAVE_ASYNC_LDS
      aReg = *(const uint4*)(asrc + k0 + TKK);
#endif
    }
#if HAVE_ASYNC_LDS
    if (it + 1 < NIT) __builtin_amdgcn_s_wait_asynccnt(1);  // current buf landed
    else              __builtin_amdgcn_s_wait_asynccnt(0);
#endif
    block_sync_lds();

    const __bf16* Ac = As[HAVE_ASYNC_LDS ? (it & 1) : 0];
    Frag a0, a1, fbg, fbu;
    const int r  = wm + l16;
    const int o1 = hi ? 8 : 0;
    a0.q[0] = *(const uint4*)&Ac[r * TKK + o1];
    a0.q[1] = *(const uint4*)&Ac[r * TKK + o1 + 16];
    a1.q[0] = *(const uint4*)&Ac[(r + 16) * TKK + o1];
    a1.q[1] = *(const uint4*)&Ac[(r + 16) * TKK + o1 + 16];
    const int nn = wn + l16;
    const int bo = hi ? 8 : 0;
    fbg.q[0] = *(const uint4*)&Bg[nn * TKK + bo];
    fbg.q[1] = *(const uint4*)&Bg[nn * TKK + bo + 16];
    fbu.q[0] = *(const uint4*)&Bu[nn * TKK + bo];
    fbu.q[1] = *(const uint4*)&Bu[nn * TKK + bo + 16];

    cg0 = wmma_bf16(a0, fbg, cg0);
    cu0 = wmma_bf16(a0, fbu, cu0);
    cg1 = wmma_bf16(a1, fbg, cg1);
    cu1 = wmma_bf16(a1, fbu, cu1);
    block_sync_lds();
  }

  // fused epilogue: h = silu(g) * u -> bf16
  const int colg = n0 + wn + l16;
#pragma unroll
  for (int r = 0; r < 8; ++r) {
    int mrow  = wm + r + (hi ? 8 : 0);
    int slot0 = m0 + mrow;
    float g = cg0[r], u = cu0[r];
    float hv = (g / (1.0f + __expf(-g))) * u;
    if (slot0 < cnt)
      hbuf[((size_t)e * NTOK + slot0) * IDIM + colg] = (__bf16)hv;
    int slot1 = slot0 + 16;
    g = cg1[r]; u = cu1[r];
    hv = (g / (1.0f + __expf(-g))) * u;
    if (slot1 < cnt)
      hbuf[((size_t)e * NTOK + slot1) * IDIM + colg] = (__bf16)hv;
  }
}

// ---------------------------------------------------------------------------
// 5) down-proj grouped GEMM + weighted RMW scatter into out.
//    block 64x128, 8 waves = 2(M) x 4(N); wave = 32x32 (4 wmma / K-step).
//    One launch per expert (sequential on stream) -> deterministic RMW.
// ---------------------------------------------------------------------------
__global__ __launch_bounds__(256) void expert_down_kernel(
    const __bf16* __restrict__ hbuf, const float* __restrict__ Wd,
    const int* __restrict__ counts, const int* __restrict__ tok_idx,
    const float* __restrict__ tok_w, float* __restrict__ out, int e) {
  const int cnt = counts[e];
  const int m0  = blockIdx.x * TM;
  if (m0 >= cnt) return;
  const int n0  = blockIdx.y * TNB;

  __shared__ __align__(16) __bf16 As[2][TM * TKK];
  __shared__ __align__(16) __bf16 Bs[TNB * TKK];    // [n][k] transposed, 8KB

  const int tid  = threadIdx.x;
  const int lane = tid & 31;
  const int wave = tid >> 5;
  const int wm   = (wave >> 2) << 5;   // 0 / 32
  const int wn   = (wave & 3) << 5;    // 0..96
  const bool hi  = lane >= 16;
  const int l16  = lane & 15;

  const int arow = tid >> 2;
  const int acol = (tid & 3) << 3;
  const __bf16* asrc = hbuf + ((size_t)e * NTOK + m0 + arow) * IDIM + acol;
  const float*  wdB  = Wd + (size_t)e * IDIM * HDIM + n0;

  // B fill: 4(k)x4(n) sub-block per thread: 8 x 32 = 256 tasks
  const int kb = (tid >> 5) << 2;   // 0..28
  const int nb = (tid & 31) << 2;   // 0..124

  float4 br0, br1, br2, br3;
#if !HAVE_ASYNC_LDS
  uint4 aReg;
#endif

  v8f c00 = {}, c01 = {}, c10 = {}, c11 = {};
  const int NIT = IDIM / TKK;

  {  // prologue prefetch
    const float* p = wdB + (size_t)kb * HDIM + nb;
    br0 = *(const float4*)p;              br1 = *(const float4*)(p + HDIM);
    br2 = *(const float4*)(p + 2 * HDIM); br3 = *(const float4*)(p + 3 * HDIM);
  }
#if HAVE_ASYNC_LDS
  __builtin_amdgcn_global_load_async_to_lds_b128(
      ASYNC_GPTR(asrc), ASYNC_LPTR(&As[0][arow * TKK + acol]), 0, 0);
#else
  aReg = *(const uint4*)asrc;
#endif

  for (int it = 0; it < NIT; ++it) {
    const int k0 = it * TKK;
#if HAVE_ASYNC_LDS
    if (it + 1 < NIT)
      __builtin_amdgcn_global_load_async_to_lds_b128(
          ASYNC_GPTR(asrc + k0 + TKK),
          ASYNC_LPTR(&As[(it + 1) & 1][arow * TKK + acol]), 0, 0);
#else
    *(uint4*)&As[0][arow * TKK + acol] = aReg;
#endif
    {
      const float* f0 = (const float*)&br0;
      const float* f1 = (const float*)&br1;
      const float* f2 = (const float*)&br2;
      const float* f3 = (const float*)&br3;
#pragma unroll
      for (int j = 0; j < 4; ++j) {
        uint2 v; v.x = pkbf(f0[j], f1[j]); v.y = pkbf(f2[j], f3[j]);
        *(uint2*)&Bs[(nb + j) * TKK + kb] = v;
      }
    }
    if (it + 1 < NIT) {
      const float* p = wdB + (size_t)(k0 + TKK + kb) * HDIM + nb;
      br0 = *(const float4*)p;              br1 = *(const float4*)(p + HDIM);
      br2 = *(const float4*)(p + 2 * HDIM); br3 = *(const float4*)(p + 3 * HDIM);
#if !HAVE_ASYNC_LDS
      aReg = *(const uint4*)(asrc + k0 + TKK);
#endif
    }
#if HAVE_ASYNC_LDS
    if (it + 1 < NIT) __builtin_amdgcn_s_wait_asynccnt(1);
    else              __builtin_amdgcn_s_wait_asynccnt(0);
#endif
    block_sync_lds();

    const __bf16* Ac = As[HAVE_ASYNC_LDS ? (it & 1) : 0];
    Frag a0, a1, b0, b1;
    const int r  = wm + l16;
    const int o1 = hi ? 8 : 0;
    a0.q[0] = *(const uint4*)&Ac[r * TKK + o1];
    a0.q[1] = *(const uint4*)&Ac[r * TKK + o1 + 16];
    a1.q[0] = *(const uint4*)&Ac[(r + 16) * TKK + o1];
    a1.q[1] = *(const uint4*)&Ac[(r + 16) * TKK + o1 + 16];
    const int nn = wn + l16;
    const int bo = hi ? 8 : 0;
    b0.q[0] = *(const uint4*)&Bs[nn * TKK + bo];
    b0.q[1] = *(const uint4*)&Bs[nn * TKK + bo + 16];
    b1.q[0] = *(const uint4*)&Bs[(nn + 16) * TKK + bo];
    b1.q[1] = *(const uint4*)&Bs[(nn + 16) * TKK + bo + 16];

    c00 = wmma_bf16(a0, b0, c00);
    c01 = wmma_bf16(a0, b1, c01);
    c10 = wmma_bf16(a1, b0, c10);
    c11 = wmma_bf16(a1, b1, c11);
    block_sync_lds();
  }

  const int col0 = n0 + wn + l16;
#pragma unroll
  for (int r = 0; r < 8; ++r) {
    const int mrow = wm + r + (hi ? 8 : 0);
    int slot = m0 + mrow;
    if (slot < cnt) {
      int   t = tok_idx[e * NTOK + slot];
      float w = tok_w[e * NTOK + slot];
      float* o = &out[(size_t)t * HDIM + col0];
      o[0]  += w * c00[r];            // race-free: token unique per expert,
      o[16] += w * c01[r];            // experts serialized by launch order
    }
    slot += 16;
    if (slot < cnt) {
      int   t = tok_idx[e * NTOK + slot];
      float w = tok_w[e * NTOK + slot];
      float* o = &out[(size_t)t * HDIM + col0];
      o[0]  += w * c10[r];
      o[16] += w * c11[r];
    }
  }
}

// ---------------------------------------------------------------------------
extern "C" void kernel_launch(void* const* d_in, const int* in_sizes, int n_in,
                              void* d_out, int out_size, void* d_ws, size_t ws_size,
                              hipStream_t stream) {
  const float* x  = (const float*)d_in[0];
  const float* Wr = (const float*)d_in[1];
  const float* Wg = (const float*)d_in[2];
  const float* Wu = (const float*)d_in[3];
  const float* Wd = (const float*)d_in[4];
  float* out = (float*)d_out;

  // workspace layout
  char* ws = (char*)d_ws;
  int*    counts  = (int*)ws;                                        // 256 B slab
  int*    tok_idx = (int*)(ws + 256);                                // E*T int
  float*  tok_w   = (float*)(ws + 256 + (size_t)NEXP * NTOK * 4);    // E*T f32
  __bf16* xb      = (__bf16*)(ws + 256 + (size_t)2 * NEXP * NTOK * 4);
  __bf16* hbuf    = xb + (size_t)NTOK * HDIM;                        // E*T*I bf16

  const int n_out = NTOK * HDIM;
  zero_init_kernel<<<(n_out / 4 + 255) / 256, 256, 0, stream>>>(out, n_out, counts);
  cvt_x_kernel<<<n_out / 4 / 256, 256, 0, stream>>>(x, xb);
  router_kernel<<<NTOK / 8, 256, 0, stream>>>(x, Wr, counts, tok_idx, tok_w);

  dim3 gA(NTOK / TM, IDIM / TN, NEXP);    // 64 x 22 x 8, early-exit on count
  expert_gateup_kernel<<<gA, 256, 0, stream>>>(xb, Wg, Wu, counts, tok_idx, hbuf);

  dim3 gB(NTOK / TM, HDIM / TNB);         // 64 x 16, one launch per expert
  for (int e = 0; e < NEXP; ++e)
    expert_down_kernel<<<gB, 256, 0, stream>>>(hbuf, Wd, counts, tok_idx, tok_w,
                                               out, e);
}
